// GlobalMixtureEncoderDiffEOL_67551245631813
// MI455X (gfx1250) — compile-verified, matrix operand order checked by reading, not run on hardware
//
#include <hip/hip_runtime.h>
#include <hip/hip_bf16.h>

// GlobalMixtureEncoderDiffEOL for MI455X (gfx1250).
// All big contractions run as v_wmma_f32_16x16x32_bf16 (f32 accumulate);
// softmax/statistics stay fp32. w_img2 (134 MB fp32) fits the 192 MB L2:
// streamed from HBM once (cvec pass), then served from L2 by the two WMMA
// passes, each of which stages+converts every w_img2 element exactly once.

#define B_  256
#define G_  1024
#define K_  512
#define E2_ 64

typedef __bf16 v16bf __attribute__((ext_vector_type(16)));
typedef float  v8f   __attribute__((ext_vector_type(8)));

struct alignas(16) U128 { unsigned int x, y, z, w; };

union FragBF {
    v16bf v;
    U128  q[2];
    unsigned short u[16];
};

__device__ __forceinline__ unsigned short f32_to_bf16(float f) {
    unsigned int u = __float_as_uint(f);
    // round-to-nearest-even
    u += 0x7FFFu + ((u >> 16) & 1u);
    return (unsigned short)(u >> 16);
}

__device__ __forceinline__ unsigned int pack_bf16(float lo, float hi) {
    return (unsigned int)f32_to_bf16(lo) | ((unsigned int)f32_to_bf16(hi) << 16);
}

__device__ __forceinline__ v8f wmma_bf16(const FragBF& a, const FragBF& b, v8f c) {
    // (neg_a, A, neg_b, B, c_mod, C, reuse_a, reuse_b)
    return __builtin_amdgcn_wmma_f32_16x16x32_bf16(false, a.v, false, b.v,
                                                   (short)0, c, false, false);
}

// ---------------------------------------------------------------------------
// Kernel 1: per-row squared norms + fp32 -> bf16 conversion of images / w_project
// ---------------------------------------------------------------------------
__global__ void __launch_bounds__(256)
prep_rows_kernel(const float* __restrict__ images, const float* __restrict__ w_project,
                 unsigned short* __restrict__ img_bf, unsigned short* __restrict__ wproj_bf,
                 float* __restrict__ xnorm, float* __restrict__ wnorm)
{
    const int rowid = blockIdx.x;
    const int tid = threadIdx.x;
    const float* src; unsigned short* dst; float* nrm; int nidx;
    if (rowid < B_) { src = images + (size_t)rowid * G_;       dst = img_bf  + (size_t)rowid * G_;       nrm = xnorm; nidx = rowid; }
    else            { int r = rowid - B_;
                      src = w_project + (size_t)r * G_;        dst = wproj_bf + (size_t)r * G_;          nrm = wnorm; nidx = r; }

    __shared__ float s_red[256];
    float a = 0.f;
    for (int j = 0; j < 4; ++j) {
        int i = tid + j * 256;
        float f = src[i];
        dst[i] = f32_to_bf16(f);
        a += f * f;
    }
    s_red[tid] = a; __syncthreads();
    for (int off = 128; off; off >>= 1) { if (tid < off) s_red[tid] += s_red[tid + off]; __syncthreads(); }
    if (tid == 0) nrm[nidx] = s_red[0];
}

// ---------------------------------------------------------------------------
// Kernel 2: xh[b,k] = -(|x_b|^2 - 2 x_b.w_k + |w_k|^2)/G  via WMMA bf16 GEMM
// Block = 64(b) x 64(k) tile; 8 waves, wave w -> mtile (w&3), ntiles {2*(w>>2),+1}
// B-matrix fragments load straight from wproj_bf (contiguous in g per lane).
// ---------------------------------------------------------------------------
__global__ void __launch_bounds__(256)
gemm_xh_kernel(const unsigned short* __restrict__ img_bf,
               const unsigned short* __restrict__ wproj_bf,
               const float* __restrict__ xnorm, const float* __restrict__ wnorm,
               float* __restrict__ xh)
{
    const int ktile = blockIdx.x;          // 8 tiles of 64 k
    const int btile = blockIdx.y;          // 4 tiles of 64 b
    const int tid = threadIdx.x, lane = tid & 31, wv = tid >> 5;
    const int mw = wv & 3, npair = wv >> 2;
    const int row  = btile * 64 + mw * 16 + (lane & 15);
    const int koff = (lane >> 4) * 8;
    const int kcol0 = ktile * 64 + npair * 32 + (lane & 15);

    v8f acc0 = {}, acc1 = {};
    for (int g0 = 0; g0 < G_; g0 += 32) {
        FragBF a;
        const unsigned short* ap = img_bf + (size_t)row * G_ + g0 + koff;
        a.q[0] = *(const U128*)ap;
        a.q[1] = *(const U128*)(ap + 16);
        FragBF b0, b1;
        const unsigned short* bp0 = wproj_bf + (size_t)kcol0 * G_ + g0 + koff;
        const unsigned short* bp1 = wproj_bf + (size_t)(kcol0 + 16) * G_ + g0 + koff;
        b0.q[0] = *(const U128*)bp0;  b0.q[1] = *(const U128*)(bp0 + 16);
        b1.q[0] = *(const U128*)bp1;  b1.q[1] = *(const U128*)(bp1 + 16);
        acc0 = wmma_bf16(a, b0, acc0);
        acc1 = wmma_bf16(a, b1, acc1);
    }
    const float invG = 1.0f / (float)G_;
    const int mrow = btile * 64 + mw * 16 + (lane >> 4) * 8;
    const int n0 = kcol0;
    for (int r = 0; r < 8; ++r) {
        int b = mrow + r;
        float xn = xnorm[b];
        xh[b * K_ + n0]      = -(xn - 2.f * acc0[r] + wnorm[n0])      * invG;
        xh[b * K_ + n0 + 16] = -(xn - 2.f * acc1[r] + wnorm[n0 + 16]) * invG;
    }
}

// ---------------------------------------------------------------------------
// Kernel 3: cvec[k,e] = sum_g w_project[k,g] * w_img2[k,g,e]
// This is the pass that streams w_img2 (134 MB) from HBM into L2.
// ---------------------------------------------------------------------------
__global__ void __launch_bounds__(256)
cvec_kernel(const float* __restrict__ w_project, const float* __restrict__ w_img2,
            float* __restrict__ cvec)
{
    const int k = blockIdx.x, tid = threadIdx.x;
    const int e = tid & 63, q = tid >> 6;
    __shared__ float s_red[256];
    const size_t base = (size_t)k * G_ * E2_;
    float acc = 0.f;
    for (int g = q; g < G_; g += 4)
        acc += w_project[(size_t)k * G_ + g] * w_img2[base + (size_t)g * E2_ + e];
    s_red[tid] = acc; __syncthreads();
    if (tid < 64) cvec[k * E2_ + tid] = s_red[tid] + s_red[tid + 64] + s_red[tid + 128] + s_red[tid + 192];
}

// ---------------------------------------------------------------------------
// Kernel 4: xe[b,k,e] = sum_g img[b,g]*w_img2[k,g,e] - cvec[k,e]   (WMMA)
// One block per k covering all 256 batch rows: wave w owns m-tiles {w, w+8}
// and all 4 n-tiles (E2=64) -> 8 f32 accum tiles per wave. The w_img2 k-slice
// is staged to LDS as bf16 transposed [e][g] (packed b32 stores), exactly once.
// ---------------------------------------------------------------------------
__global__ void __launch_bounds__(256)
xe_gemm_kernel(const unsigned short* __restrict__ img_bf,
               const float* __restrict__ w_img2,
               const float* __restrict__ cvec, float* __restrict__ xe)
{
    const int k = blockIdx.x;
    __shared__ __align__(16) unsigned short lds_t[64 * 32];   // [e][g]
    const int tid = threadIdx.x, lane = tid & 31, wv = tid >> 5;
    const int m0   = wv * 16 + (lane & 15);
    const int m1   = (wv + 8) * 16 + (lane & 15);
    const int koff = (lane >> 4) * 8;
    const size_t w2base = (size_t)k * G_ * E2_;

    v8f acc[8] = {};
    for (int g0 = 0; g0 < G_; g0 += 32) {
        __syncthreads();
        {   // stage [e=64][g=32] tile, two g per thread-step packed into b32
            const int e = tid & 63, gq = tid >> 6;
            unsigned int* ldsw = (unsigned int*)lds_t;
            for (int j = 0; j < 4; ++j) {
                int gg = (gq + j * 4) * 2;
                float f0 = w_img2[w2base + (size_t)(g0 + gg) * E2_ + e];
                float f1 = w_img2[w2base + (size_t)(g0 + gg + 1) * E2_ + e];
                ldsw[(e * 32 + gg) >> 1] = pack_bf16(f0, f1);
            }
        }
        __syncthreads();
        FragBF a0, a1;
        const unsigned short* ap0 = img_bf + (size_t)m0 * G_ + g0 + koff;
        const unsigned short* ap1 = img_bf + (size_t)m1 * G_ + g0 + koff;
        a0.q[0] = *(const U128*)ap0;  a0.q[1] = *(const U128*)(ap0 + 16);
        a1.q[0] = *(const U128*)ap1;  a1.q[1] = *(const U128*)(ap1 + 16);
        for (int nt = 0; nt < 4; ++nt) {
            const int e = nt * 16 + (lane & 15);
            FragBF b;
            b.q[0] = *(const U128*)&lds_t[e * 32 + koff];
            b.q[1] = *(const U128*)&lds_t[e * 32 + koff + 16];
            acc[nt]     = wmma_bf16(a0, b, acc[nt]);
            acc[nt + 4] = wmma_bf16(a1, b, acc[nt + 4]);
        }
    }
    for (int half = 0; half < 2; ++half) {
        const int mrow = (wv + half * 8) * 16 + (lane >> 4) * 8;
        for (int nt = 0; nt < 4; ++nt) {
            const int e = nt * 16 + (lane & 15);
            const float cv = cvec[k * E2_ + e];
            for (int r = 0; r < 8; ++r)
                xe[((size_t)(mrow + r) * K_ + k) * E2_ + e] = acc[half * 4 + nt][r] - cv;
        }
    }
}

// ---------------------------------------------------------------------------
// Kernel 5: per-row softmax(xh) -> xp (LDS only); lat1 = xp@wrec; lat2 = sum_k xp*xe
// ---------------------------------------------------------------------------
__global__ void __launch_bounds__(256)
enc_reduce_kernel(const float* __restrict__ xh, const float* __restrict__ w_rec,
                  const float* __restrict__ xe,
                  float* __restrict__ lat1, float* __restrict__ lat2)
{
    const int b = blockIdx.x, tid = threadIdx.x;
    __shared__ float s_red[256];
    __shared__ float s_xp[K_];
    float v0 = xh[b * K_ + tid], v1 = xh[b * K_ + 256 + tid];
    s_red[tid] = fmaxf(v0, v1); __syncthreads();
    for (int off = 128; off; off >>= 1) { if (tid < off) s_red[tid] = fmaxf(s_red[tid], s_red[tid + off]); __syncthreads(); }
    float mx = s_red[0]; __syncthreads();
    float e0 = __expf(v0 - mx), e1 = __expf(v1 - mx);
    s_red[tid] = e0 + e1; __syncthreads();
    for (int off = 128; off; off >>= 1) { if (tid < off) s_red[tid] += s_red[tid + off]; __syncthreads(); }
    float inv = 1.0f / s_red[0]; __syncthreads();
    s_xp[tid] = e0 * inv; s_xp[tid + 256] = e1 * inv;
    __syncthreads();

    const int e = tid & 63, q = tid >> 6;
    float a1 = 0.f;
    for (int k = q * 128; k < q * 128 + 128; ++k) a1 += s_xp[k] * w_rec[e * K_ + k];
    s_red[tid] = a1; __syncthreads();
    if (tid < 64) lat1[b * E2_ + tid] = s_red[tid] + s_red[tid + 64] + s_red[tid + 128] + s_red[tid + 192];
    __syncthreads();
    float a2 = 0.f;
    for (int k = q * 128; k < q * 128 + 128; ++k) a2 += s_xp[k] * xe[((size_t)b * K_ + k) * E2_ + e];
    s_red[tid] = a2; __syncthreads();
    if (tid < 64) lat2[b * E2_ + tid] = s_red[tid] + s_red[tid + 64] + s_red[tid + 128] + s_red[tid + 192];
}

// ---------------------------------------------------------------------------
// Kernel 6: decode distances + softmax -> xp2; emit xp2_bf and t[b,k,e]=xp2*lat2 (bf16)
// ---------------------------------------------------------------------------
__global__ void __launch_bounds__(256)
dec_softmax_kernel(const float* __restrict__ lat1, const float* __restrict__ lat2,
                   const float* __restrict__ w_rec,
                   unsigned short* __restrict__ xp2_bf, unsigned short* __restrict__ t_bf)
{
    const int b = blockIdx.x, tid = threadIdx.x;
    __shared__ float s_l1[E2_], s_l2[E2_];
    __shared__ float s_h[K_];
    __shared__ float s_red[256];
    if (tid < 64) { s_l1[tid] = lat1[b * E2_ + tid]; s_l2[tid] = lat2[b * E2_ + tid]; }
    __syncthreads();
    for (int j = 0; j < 2; ++j) {
        const int k = tid + j * 256;
        float d = 0.f;
        for (int e = 0; e < E2_; ++e) { float t = s_l1[e] - w_rec[e * K_ + k]; d += t * t; }
        s_h[k] = -d * (1.0f / (float)E2_);
    }
    __syncthreads();
    float v0 = s_h[tid], v1 = s_h[tid + 256];
    s_red[tid] = fmaxf(v0, v1); __syncthreads();
    for (int off = 128; off; off >>= 1) { if (tid < off) s_red[tid] = fmaxf(s_red[tid], s_red[tid + off]); __syncthreads(); }
    float mx = s_red[0]; __syncthreads();
    float e0 = __expf(v0 - mx), e1 = __expf(v1 - mx);
    s_red[tid] = e0 + e1; __syncthreads();
    for (int off = 128; off; off >>= 1) { if (tid < off) s_red[tid] += s_red[tid + off]; __syncthreads(); }
    float inv = 1.0f / s_red[0]; __syncthreads();
    float p0 = e0 * inv, p1 = e1 * inv;
    s_h[tid] = p0; s_h[tid + 256] = p1;
    xp2_bf[b * K_ + tid]       = f32_to_bf16(p0);
    xp2_bf[b * K_ + tid + 256] = f32_to_bf16(p1);
    __syncthreads();
    for (int idx = tid; idx < K_ * E2_; idx += 256) {
        const int k = idx >> 6, e = idx & 63;
        t_bf[((size_t)b * K_ + k) * E2_ + e] = f32_to_bf16(s_h[k] * s_l2[e]);
    }
}

// ---------------------------------------------------------------------------
// Kernel 7: out[b,g] = sum_k xp2[b,k]*w_project[k,g]   (WMMA, plain store)
// wproj tile transposed into LDS so B-fragments are contiguous in k.
// ---------------------------------------------------------------------------
__global__ void __launch_bounds__(256)
recon_base_kernel(const unsigned short* __restrict__ xp2_bf,
                  const unsigned short* __restrict__ wproj_bf,
                  float* __restrict__ out)
{
    const int gt = blockIdx.x, bt = blockIdx.y;
    __shared__ __align__(16) unsigned short lds_t[64 * 32];   // [g][k]
    const int tid = threadIdx.x, lane = tid & 31, wv = tid >> 5;
    const int mw = wv & 3, npair = wv >> 2;
    const int g0 = gt * 64, b0 = bt * 64;
    const int row  = b0 + mw * 16 + (lane & 15);
    const int koff = (lane >> 4) * 8;

    v8f acc0 = {}, acc1 = {};
    for (int k0 = 0; k0 < K_; k0 += 32) {
        __syncthreads();
        {
            const int g = tid & 63, kq = tid >> 6;
            for (int j = 0; j < 8; ++j) {
                int kl = kq + j * 4;
                lds_t[g * 32 + kl] = wproj_bf[(size_t)(k0 + kl) * G_ + g0 + g];
            }
        }
        __syncthreads();
        FragBF a;
        const unsigned short* ap = xp2_bf + (size_t)row * K_ + k0 + koff;
        a.q[0] = *(const U128*)ap;
        a.q[1] = *(const U128*)(ap + 16);
        const int gl0 = npair * 32 + (lane & 15);
        FragBF bf0, bf1;
        bf0.q[0] = *(const U128*)&lds_t[gl0 * 32 + koff];
        bf0.q[1] = *(const U128*)&lds_t[gl0 * 32 + koff + 16];
        bf1.q[0] = *(const U128*)&lds_t[(gl0 + 16) * 32 + koff];
        bf1.q[1] = *(const U128*)&lds_t[(gl0 + 16) * 32 + koff + 16];
        acc0 = wmma_bf16(a, bf0, acc0);
        acc1 = wmma_bf16(a, bf1, acc1);
    }
    const int mrow = b0 + mw * 16 + (lane >> 4) * 8;
    const int n0 = g0 + npair * 32 + (lane & 15);
    for (int r = 0; r < 8; ++r) {
        out[(size_t)(mrow + r) * G_ + n0]      = acc0[r];
        out[(size_t)(mrow + r) * G_ + n0 + 16] = acc1[r];
    }
}

// ---------------------------------------------------------------------------
// Kernel 8: out[b,g] += sum_{k,e} t[b,k,e]*w_img2[k,g,e]   (WMMA, atomic add)
// Grid (16 g-tiles, 16 k-chunks of 32); each block covers all 256 b rows
// (wave w -> m-tiles {w, w+8}, 8 accum tiles), so each w_img2 element is
// staged+converted exactly once across the grid. Staging packs e-pairs
// (contiguous fp32) into b32 LDS stores. Cross-chunk accumulation via
// global_atomic_add_f32.
// ---------------------------------------------------------------------------
__global__ void __launch_bounds__(256)
xd2_gemm_kernel(const unsigned short* __restrict__ t_bf,
                const float* __restrict__ w_img2, float* __restrict__ out)
{
    const int gt = blockIdx.x;   // 16 g-tiles of 64
    const int kc = blockIdx.y;   // 16 chunks of 32 k
    __shared__ __align__(16) unsigned short lds_t[64 * 64];   // [g][e]
    const int tid = threadIdx.x, lane = tid & 31, wv = tid >> 5;
    const int g0 = gt * 64;
    const int koff = (lane >> 4) * 8;
    const int row0 = wv * 16 + (lane & 15);
    const int row1 = (wv + 8) * 16 + (lane & 15);

    v8f acc[8] = {};
    for (int k = kc * 32; k < kc * 32 + 32; ++k) {
        __syncthreads();
        {   // stage [g=64][e=64] tile; e-pairs are contiguous fp32 -> packed b32
            const int el2 = (tid & 31) * 2, gq = tid >> 5;
            const size_t base = ((size_t)k * G_ + g0) * E2_;
            unsigned int* ldsw = (unsigned int*)lds_t;
            for (int j = 0; j < 8; ++j) {
                int gl = gq + j * 8;
                const float* sp = &w_img2[base + (size_t)gl * E2_ + el2];
                ldsw[(gl * 64 + el2) >> 1] = pack_bf16(sp[0], sp[1]);
            }
        }
        __syncthreads();
        for (int es = 0; es < 64; es += 32) {
            FragBF a0, a1;
            const unsigned short* ap0 = t_bf + ((size_t)row0 * K_ + k) * E2_ + es + koff;
            const unsigned short* ap1 = t_bf + ((size_t)row1 * K_ + k) * E2_ + es + koff;
            a0.q[0] = *(const U128*)ap0;  a0.q[1] = *(const U128*)(ap0 + 16);
            a1.q[0] = *(const U128*)ap1;  a1.q[1] = *(const U128*)(ap1 + 16);
            for (int nt = 0; nt < 4; ++nt) {
                const int gl = nt * 16 + (lane & 15);
                FragBF b;
                b.q[0] = *(const U128*)&lds_t[gl * 64 + es + koff];
                b.q[1] = *(const U128*)&lds_t[gl * 64 + es + koff + 16];
                acc[nt]     = wmma_bf16(a0, b, acc[nt]);
                acc[nt + 4] = wmma_bf16(a1, b, acc[nt + 4]);
            }
        }
    }
    for (int half = 0; half < 2; ++half) {
        const int mrow = (wv + half * 8) * 16 + (lane >> 4) * 8;
        for (int nt = 0; nt < 4; ++nt) {
            const int n = g0 + nt * 16 + (lane & 15);
            for (int r = 0; r < 8; ++r)
                atomicAdd(&out[(size_t)(mrow + r) * G_ + n], acc[half * 4 + nt][r]);
        }
    }
}

// ---------------------------------------------------------------------------
extern "C" void kernel_launch(void* const* d_in, const int* in_sizes, int n_in,
                              void* d_out, int out_size, void* d_ws, size_t ws_size,
                              hipStream_t stream)
{
    (void)in_sizes; (void)n_in; (void)out_size; (void)ws_size;
    const float* images    = (const float*)d_in[0];
    const float* w_project = (const float*)d_in[1];
    const float* w_rec     = (const float*)d_in[2];
    const float* w_img2    = (const float*)d_in[3];
    float* out = (float*)d_out;

    char* w = (char*)d_ws;
    auto alloc = [&](size_t bytes) { void* p = (void*)w; w += (bytes + 255) & ~(size_t)255; return p; };
    float* xh    = (float*)alloc((size_t)B_ * K_ * 4);
    float* lat1  = (float*)alloc((size_t)B_ * E2_ * 4);
    float* lat2  = (float*)alloc((size_t)B_ * E2_ * 4);
    float* cvec  = (float*)alloc((size_t)K_ * E2_ * 4);
    float* xnorm = (float*)alloc((size_t)B_ * 4);
    float* wnorm = (float*)alloc((size_t)K_ * 4);
    float* xe    = (float*)alloc((size_t)B_ * K_ * E2_ * 4);          // 32 MB
    unsigned short* img_bf   = (unsigned short*)alloc((size_t)B_ * G_ * 2);
    unsigned short* wproj_bf = (unsigned short*)alloc((size_t)K_ * G_ * 2);
    unsigned short* xp2_bf   = (unsigned short*)alloc((size_t)B_ * K_ * 2);
    unsigned short* t_bf     = (unsigned short*)alloc((size_t)B_ * K_ * E2_ * 2);   // 16 MB

    prep_rows_kernel<<<B_ + K_, 256, 0, stream>>>(images, w_project, img_bf, wproj_bf, xnorm, wnorm);
    gemm_xh_kernel<<<dim3(K_ / 64, B_ / 64), 256, 0, stream>>>(img_bf, wproj_bf, xnorm, wnorm, xh);
    cvec_kernel<<<K_, 256, 0, stream>>>(w_project, w_img2, cvec);
    xe_gemm_kernel<<<K_, 256, 0, stream>>>(img_bf, w_img2, cvec, xe);
    enc_reduce_kernel<<<B_, 256, 0, stream>>>(xh, w_rec, xe, lat1, lat2);
    dec_softmax_kernel<<<B_, 256, 0, stream>>>(lat1, lat2, w_rec, xp2_bf, t_bf);
    recon_base_kernel<<<dim3(G_ / 64, B_ / 64), 256, 0, stream>>>(xp2_bf, wproj_bf, out);
    xd2_gemm_kernel<<<dim3(G_ / 64, 16), 256, 0, stream>>>(t_bf, w_img2, out);
}
